// PhraseModel_41764261986402
// MI455X (gfx1250) — matrix operand order repaired and checked by reading, not compiled, not run on hardware
//
#include <hip/hip_runtime.h>
#include <hip/hip_bf16.h>
#include <stddef.h>

// ---------------------------------------------------------------------------
// Problem constants (from the reference):  B=4096, D=510, K=128, P=332
// ---------------------------------------------------------------------------
#define DIM      510
#define DIMP     512           // D padded to a multiple of 32 (WMMA K-step)
#define DPAD     520           // LDS row stride (+16B) to break bank alignment
#define NCODE    128
#define ROWS     32            // rows per block (each row processed for z AND z_pre)
#define THREADS  128           // 4 wave32s: wave = (input, code-half)

typedef __attribute__((ext_vector_type(16))) __bf16 v16bf;
typedef __attribute__((ext_vector_type(8)))  __bf16 v8bf;
typedef __attribute__((ext_vector_type(8)))  float  v8f;

// ---------------------------------------------------------------------------
// Prep: codebook f32[128][510] -> bf16[128][512] (zero-padded) + f32 ||c||^2.
// One block per code. Rebuilt every call (workspace is poisoned by harness).
// ---------------------------------------------------------------------------
__global__ __launch_bounds__(256)
void vq_prep_kernel(const float* __restrict__ codebook,
                    unsigned short* __restrict__ cb_bf16_raw,
                    float* __restrict__ cnorm)
{
    __bf16* cb = (__bf16*)cb_bf16_raw;
    const int k   = blockIdx.x;
    const int tid = threadIdx.x;
    __shared__ float red[256];

    float s = 0.0f;
    for (int d = tid; d < DIMP; d += 256) {
        float v = (d < DIM) ? codebook[(size_t)k * DIM + d] : 0.0f;
        cb[(size_t)k * DIMP + d] = (__bf16)v;
        s += v * v;
    }
    red[tid] = s;
    __syncthreads();
    for (int off = 128; off > 0; off >>= 1) {
        if (tid < off) red[tid] += red[tid + off];
        __syncthreads();
    }
    if (tid == 0) cnorm[k] = red[0];
}

// ---------------------------------------------------------------------------
// Main fused kernel, per block of 32 rows (z AND z_pre):
//   1) stage z / z_pre rows to LDS as padded bf16 (coalesced f32 reads)
//   2) WMMA bf16 score tiles:  zc = Z_tile @ C^T  (f32 accum)
//      wave tile = 32 rows x 64 codes -> each B fragment feeds 2 WMMAs
//      (halves L2 codebook traffic vs a 16-row wave tile)
//   3) argmin_k ( ||c_k||^2 - 2*zc )  via shfl_xor + LDS merge
//   4) out = codebook[k1] + codebook[k2] + phrase_table[pos]  (coalesced)
// ---------------------------------------------------------------------------
__global__ __launch_bounds__(THREADS)
void vq_main_kernel(const float* __restrict__ z,
                    const float* __restrict__ z_pre,
                    const unsigned short* __restrict__ cb_bf16_raw,
                    const float* __restrict__ cnorm,
                    const float* __restrict__ codebook_f32,
                    const float* __restrict__ phrase,
                    const int*   __restrict__ pos,
                    float*       __restrict__ out)
{
    const __bf16* cb = (const __bf16*)cb_bf16_raw;

    __shared__ __align__(16) __bf16 sA[2][ROWS][DPAD];   // ~65 KB staged inputs
    __shared__ float s_pmin[2][2][ROWS];                 // [input][codehalf][row]
    __shared__ int   s_pidx[2][2][ROWS];
    __shared__ int   s_k[2][ROWS];                       // final argmin per row

    const int tid = threadIdx.x;
    const int m0  = blockIdx.x * ROWS;

    // ---- 1) stage inputs to LDS as bf16, padded 510 -> 512 -----------------
    for (int idx = tid; idx < 2 * ROWS * DIMP; idx += THREADS) {
        const int inp = idx / (ROWS * DIMP);
        const int rem = idx - inp * (ROWS * DIMP);
        const int r   = rem >> 9;           // / DIMP
        const int d   = rem & (DIMP - 1);   // % DIMP
        const float* src = inp ? z_pre : z;
        float v = (d < DIM) ? src[(size_t)(m0 + r) * DIM + d] : 0.0f;
        sA[inp][r][d] = (__bf16)v;
    }
    __syncthreads();

    // ---- 2) WMMA score tiles ----------------------------------------------
    const int wid   = tid >> 5;        // wave32 id: 0..3
    const int lane  = tid & 31;
    const int lrow  = lane & 15;
    const int half  = lane >> 4;

    const int inp     = wid & 1;        // z or z_pre
    const int colhalf = wid >> 1;       // codes [colhalf*64, +64)

    // acc[s][t]: rows [s*16, +16), codes [colhalf*64 + t*16, +16)
    v8f acc[2][4] = {{v8f{}, v8f{}, v8f{}, v8f{}},
                     {v8f{}, v8f{}, v8f{}, v8f{}}};

    const __bf16* arow0 = &sA[inp][lrow][0];
    const __bf16* arow1 = &sA[inp][16 + lrow][0];

    for (int d0 = 0; d0 < DIMP; d0 += 32) {
        // A fragments (16x32 bf16): lanes 0-15 -> K = d0+{0..15},
        //                           lanes 16-31 -> K = d0+{16..31}
        v8bf a00 = *(const v8bf*)(arow0 + d0 + half * 16);
        v8bf a01 = *(const v8bf*)(arow0 + d0 + 8 + half * 16);
        v8bf a10 = *(const v8bf*)(arow1 + d0 + half * 16);
        v8bf a11 = *(const v8bf*)(arow1 + d0 + 8 + half * 16);
        v16bf a0 = __builtin_shufflevector(a00, a01,
                       0, 1, 2, 3, 4, 5, 6, 7, 8, 9, 10, 11, 12, 13, 14, 15);
        v16bf a1 = __builtin_shufflevector(a10, a11,
                       0, 1, 2, 3, 4, 5, 6, 7, 8, 9, 10, 11, 12, 13, 14, 15);
#pragma unroll
        for (int t = 0; t < 4; ++t) {
            // B fragment (32x16 bf16, column-major load): lane holds one
            // codebook row (== one B column), 16 contiguous K values.
            // Reused by both row strips -> 2 WMMAs per fragment.
            const int code = colhalf * 64 + t * 16 + lrow;
            v16bf b = *(const v16bf*)(cb + (size_t)code * DIMP + d0 + half * 16);
            acc[0][t] = __builtin_amdgcn_wmma_f32_16x16x32_bf16(
                            false, a0, false, b, (short)0, acc[0][t], false, false);
            acc[1][t] = __builtin_amdgcn_wmma_f32_16x16x32_bf16(
                            false, a1, false, b, (short)0, acc[1][t], false, false);
        }
    }

    // ---- 3) argmin over this wave's 64 codes, both 16-row strips ----------
    // acc[s][t] element r on this lane = zc[row m0+s*16 + r + 8*half][code]
#pragma unroll
    for (int s = 0; s < 2; ++s) {
        float vmin[8];
        int   imin[8];
#pragma unroll
        for (int r = 0; r < 8; ++r) { vmin[r] = __FLT_MAX__; imin[r] = 0x7fffffff; }

#pragma unroll
        for (int t = 0; t < 4; ++t) {
            const int   code = colhalf * 64 + t * 16 + lrow;
            const float cn   = cnorm[code];
#pragma unroll
            for (int r = 0; r < 8; ++r) {
                float dist = cn - 2.0f * acc[s][t][r];
                if (dist < vmin[r] || (dist == vmin[r] && code < imin[r])) {
                    vmin[r] = dist;
                    imin[r] = code;
                }
            }
        }
        // reduce across the 16 lanes sharing each row (stays in half-wave)
#pragma unroll
        for (int m = 1; m <= 8; m <<= 1) {
#pragma unroll
            for (int r = 0; r < 8; ++r) {
                float ov = __shfl_xor(vmin[r], m, 32);
                int   oi = __shfl_xor(imin[r], m, 32);
                if (ov < vmin[r] || (ov == vmin[r] && oi < imin[r])) {
                    vmin[r] = ov;
                    imin[r] = oi;
                }
            }
        }
        if (lrow == 0) {
#pragma unroll
            for (int r = 0; r < 8; ++r) {
                const int rl = s * 16 + 8 * half + r;
                s_pmin[inp][colhalf][rl] = vmin[r];
                s_pidx[inp][colhalf][rl] = imin[r];
            }
        }
    }
    __syncthreads();

    // merge the two code-halves -> final index per (input,row)
    if (tid < 2 * ROWS) {
        const int inp2 = tid >> 5;   // ROWS == 32
        const int rl   = tid & 31;
        float v0 = s_pmin[inp2][0][rl]; int i0 = s_pidx[inp2][0][rl];
        float v1 = s_pmin[inp2][1][rl]; int i1 = s_pidx[inp2][1][rl];
        s_k[inp2][rl] = (v1 < v0 || (v1 == v0 && i1 < i0)) ? i1 : i0;
    }
    __syncthreads();

    // ---- 4) fused gather + add + coalesced store --------------------------
    for (int rl = 0; rl < ROWS; ++rl) {
        const int k1 = s_k[0][rl];
        const int k2 = s_k[1][rl];
        const int p  = pos[m0 + rl];
        const float* c1 = codebook_f32 + (size_t)k1 * DIM;
        const float* c2 = codebook_f32 + (size_t)k2 * DIM;
        const float* ph = phrase       + (size_t)p  * DIM;
        float*       o  = out          + (size_t)(m0 + rl) * DIM;
        for (int d = tid; d < DIM; d += THREADS)
            o[d] = c1[d] + c2[d] + ph[d];
    }
}

// ---------------------------------------------------------------------------
// Host entry
//   d_in: z[4096*510] f32, z_pre[4096*510] f32, codebook[128*510] f32,
//         phrase_table[332*510] f32, position_number[4096] int
//   d_out: [4096*510] f32
// ---------------------------------------------------------------------------
extern "C" void kernel_launch(void* const* d_in, const int* in_sizes, int n_in,
                              void* d_out, int out_size, void* d_ws, size_t ws_size,
                              hipStream_t stream)
{
    const float* z        = (const float*)d_in[0];
    const float* z_pre    = (const float*)d_in[1];
    const float* codebook = (const float*)d_in[2];
    const float* phrase   = (const float*)d_in[3];
    const int*   pos      = (const int*)d_in[4];
    float*       out      = (float*)d_out;

    // workspace layout: bf16 codebook [128][512], then f32 cnorm[128]
    unsigned short* cb_bf16 = (unsigned short*)d_ws;
    float* cnorm = (float*)((char*)d_ws + (size_t)NCODE * DIMP * sizeof(unsigned short));

    vq_prep_kernel<<<NCODE, 256, 0, stream>>>(codebook, cb_bf16, cnorm);

    const int nblocks = 4096 / ROWS;   // 128 blocks x 4 waves
    vq_main_kernel<<<nblocks, THREADS, 0, stream>>>(
        z, z_pre, cb_bf16, cnorm, codebook, phrase, pos, out);
}
